// InterpolationStage_56212531970314
// MI455X (gfx1250) — matrix operand order repaired and checked by reading, not compiled, not run on hardware
//
#include <hip/hip_runtime.h>
#include <hip/hip_bf16.h>
#include <math.h>

// ---------------------------------------------------------------------------
// Types
// ---------------------------------------------------------------------------
typedef __bf16 bf16_t;
typedef __attribute__((ext_vector_type(16))) __bf16 v16bf;
typedef __attribute__((ext_vector_type(8)))  __bf16 v8bf;
typedef __attribute__((ext_vector_type(4)))  __bf16 v4bf;
typedef __attribute__((ext_vector_type(8)))  float  v8f;

#define N_ENC 32768
#define N_DEC 8192
#define D     256
#define KNBR  16

// ---------------------------------------------------------------------------
// Precompute kernels: fold weights so the keys/vals GEMMs vanish.
// All folded weights are written TRANSPOSED (Bt[n][k]) so WMMA B-fragments
// are two contiguous global_load_b128 per lane, no LDS staging needed.
// ---------------------------------------------------------------------------

// Wqk_t[j][i] = sum_d Wq[i][d] * Wk[j][d]   (= (Wq @ Wk^T) transposed)
__global__ __launch_bounds__(256) void prep_wqk(const float* __restrict__ Wq,
                                                const float* __restrict__ Wk,
                                                bf16_t* __restrict__ Wqk_t) {
  const int i = blockIdx.x;      // GEMM K dim
  const int j = threadIdx.x;     // GEMM N dim
  float s = 0.f;
  const float* wqr = Wq + i * D;
  const float* wkr = Wk + j * D;
  for (int d = 0; d < D; ++d) s = fmaf(wqr[d], wkr[d], s);
  Wqk_t[j * D + i] = (bf16_t)s;
}

// Wv1_t[o][i] = sum_j Wv[i][j] * W1[j][o]   (= (Wv @ W1_top) transposed)
__global__ __launch_bounds__(256) void prep_wv1(const float* __restrict__ Wv,
                                                const float* __restrict__ W1,
                                                bf16_t* __restrict__ Wv1_t) {
  const int i = blockIdx.x;
  const int o = threadIdx.x;
  float s = 0.f;
  const float* wvr = Wv + i * D;
  for (int j = 0; j < D; ++j) s = fmaf(wvr[j], W1[j * D + o], s);
  Wv1_t[o * D + i] = (bf16_t)s;
}

// Small vectors: qb = bq @ Wk^T ; vvec = Wq @ bk ; b1p = b1 + bv @ W1_top ; c = bq.bk
__global__ __launch_bounds__(256) void prep_vec(const float* __restrict__ bq,
                                                const float* __restrict__ Wk,
                                                const float* __restrict__ Wq,
                                                const float* __restrict__ bk,
                                                const float* __restrict__ bv,
                                                const float* __restrict__ W1,
                                                const float* __restrict__ b1,
                                                float* __restrict__ qb,
                                                float* __restrict__ vvec,
                                                float* __restrict__ b1p,
                                                float* __restrict__ cbuf) {
  const int t = threadIdx.x;
  if (blockIdx.x == 0) {
    float s = 0.f;
    for (int d = 0; d < D; ++d) s = fmaf(bq[d], Wk[t * D + d], s);
    qb[t] = s;
  } else if (blockIdx.x == 1) {
    float s = 0.f;
    for (int d = 0; d < D; ++d) s = fmaf(Wq[t * D + d], bk[d], s);
    vvec[t] = s;
  } else if (blockIdx.x == 2) {
    float s = b1[t];
    for (int j = 0; j < D; ++j) s = fmaf(bv[j], W1[j * D + t], s);
    b1p[t] = s;
  } else if (t == 0) {
    float s = 0.f;
    for (int d = 0; d < D; ++d) s = fmaf(bq[d], bk[d], s);
    cbuf[0] = s;
  }
}

// Transposed bf16 copies of W1 bottom half and W2:
//   W1b_t[n][k] = W1[(256+k)][n],  W2_t[n][k] = W2[k][n]
__global__ __launch_bounds__(256) void prep_cvt(const float* __restrict__ W1,
                                                const float* __restrict__ W2,
                                                bf16_t* __restrict__ W1b_t,
                                                bf16_t* __restrict__ W2_t) {
  const int x = blockIdx.x * blockDim.x + threadIdx.x;  // 0..65535
  const int n = x >> 8, k = x & 255;
  W1b_t[n * D + k] = (bf16_t)W1[(D + k) * D + n];
  W2_t[n * D + k]  = (bf16_t)W2[k * D + n];
}

// encoder_features f32 -> bf16 (row-major, A-operand for K1/K3).
__global__ __launch_bounds__(256) void prep_enc(const float* __restrict__ enc,
                                                bf16_t* __restrict__ enc_bf) {
  const size_t i = ((size_t)blockIdx.x * blockDim.x + threadIdx.x) * 4;
  const float4 v = *(const float4*)(enc + i);
  v4bf w;
  w[0] = (bf16_t)v.x; w[1] = (bf16_t)v.y; w[2] = (bf16_t)v.z; w[3] = (bf16_t)v.w;
  *(v4bf*)(enc_bf + i) = w;
}

// Generic 4-byte copy (pos / labels passthrough into d_out tail).
__global__ void copy_u32(const unsigned* __restrict__ src,
                         unsigned* __restrict__ dst, int n) {
  int i = blockIdx.x * blockDim.x + threadIdx.x;
  if (i < n) dst[i] = src[i];
}

// ---------------------------------------------------------------------------
// Direct-from-global WMMA GEMM (no LDS, no barriers in the main loop).
//   Out[M,256] = epilogue( A1@B1t^T (+ A2@B2t^T) + bias )
// A: bf16 row-major [M][256]. Bt: bf16 [N][K] (pre-transposed weights).
// 256 threads = 8 waves; wave (wm=w>>2, wn=w&3) computes MT*16 rows x 64 cols
// (MT x 4 WMMA tiles). Per 32-wide k-step: MT*2 + 8 global_load_b128,
// MT*4 v_wmma_f32_16x16x32_bf16. Weight reuse served by WGP$/L2.
// ---------------------------------------------------------------------------

template <int MT>
__device__ inline void panel(const bf16_t* __restrict__ A,
                             const bf16_t* __restrict__ Bt,
                             int wrow0, int wn, int lane, v8f acc[MT][4]) {
  const int lm = lane & 15, lh = lane >> 4;
  const bf16_t* ap[MT];
#pragma unroll
  for (int mt = 0; mt < MT; ++mt)
    ap[mt] = A + (size_t)(wrow0 + mt * 16 + lm) * D + lh * 8;
  const bf16_t* bp[4];
#pragma unroll
  for (int nt = 0; nt < 4; ++nt)
    bp[nt] = Bt + (size_t)(wn * 64 + nt * 16 + lm) * D + lh * 8;

#pragma unroll 2
  for (int k0 = 0; k0 < D; k0 += 32) {
    v16bf af[MT], bfg[4];
#pragma unroll
    for (int mt = 0; mt < MT; ++mt) {
      // A 16x32 fragment: lane<16 -> K 0..7 & 16..23 ; lane>=16 -> +8
      const v8bf lo = *(const v8bf*)(ap[mt] + k0);
      const v8bf hi = *(const v8bf*)(ap[mt] + k0 + 16);
#pragma unroll
      for (int i = 0; i < 8; ++i) { af[mt][i] = lo[i]; af[mt][8 + i] = hi[i]; }
    }
#pragma unroll
    for (int nt = 0; nt < 4; ++nt) {
      // B 32x16 fragment: lane<16 -> K 0..7 & 16..23 ; lane>=16 -> +8
      const v8bf lo = *(const v8bf*)(bp[nt] + k0);
      const v8bf hi = *(const v8bf*)(bp[nt] + k0 + 16);
#pragma unroll
      for (int i = 0; i < 8; ++i) { bfg[nt][i] = lo[i]; bfg[nt][8 + i] = hi[i]; }
    }
#pragma unroll
    for (int mt = 0; mt < MT; ++mt)
#pragma unroll
      for (int nt = 0; nt < 4; ++nt)
        acc[mt][nt] = __builtin_amdgcn_wmma_f32_16x16x32_bf16(
            false, af[mt], false, bfg[nt], (short)0, acc[mt][nt], false, false);
  }
}

template <int MT, bool HAS_A2, bool RELU, bool LN, typename TO>
__global__ __launch_bounds__(256) void gemm256_kernel(
    const bf16_t* __restrict__ A1, const bf16_t* __restrict__ A2,
    const bf16_t* __restrict__ B1t, const bf16_t* __restrict__ B2t,
    const float* __restrict__ bias,
    const float* __restrict__ gamma, const float* __restrict__ beta,
    TO* __restrict__ Out) {
  __shared__ float Hs[LN ? 32 * 257 : 1];
  __shared__ float mu_s[LN ? 32 : 1];
  __shared__ float rs_s[LN ? 32 : 1];

  const int t = threadIdx.x;
  const int lane = t & 31;
  const int w = t >> 5, wm = w >> 2, wn = w & 3;
  const int block_rows = 2 * MT * 16;
  const int row0 = blockIdx.x * block_rows;
  const int wrow0 = row0 + wm * MT * 16;

  const v8f zero = {0.f, 0.f, 0.f, 0.f, 0.f, 0.f, 0.f, 0.f};
  v8f acc[MT][4];
#pragma unroll
  for (int mt = 0; mt < MT; ++mt)
#pragma unroll
    for (int nt = 0; nt < 4; ++nt) acc[mt][nt] = zero;

  panel<MT>(A1, B1t, wrow0, wn, lane, acc);
  if constexpr (HAS_A2) panel<MT>(A2, B2t, wrow0, wn, lane, acc);

  // ---- epilogue: bias (+relu) (+layernorm) ----
#pragma unroll
  for (int mt = 0; mt < MT; ++mt) {
#pragma unroll
    for (int nt = 0; nt < 4; ++nt) {
      const int col = wn * 64 + nt * 16 + (lane & 15);
      const float bv = bias[col];
#pragma unroll
      for (int r = 0; r < 8; ++r) {
        const int rlocal = wm * MT * 16 + mt * 16 + (lane >> 4) * 8 + r;
        float val = acc[mt][nt][r] + bv;
        if constexpr (RELU) val = fmaxf(val, 0.f);
        if constexpr (LN) {
          Hs[rlocal * 257 + col] = val;
        } else {
          Out[(size_t)(row0 + rlocal) * D + col] = (TO)val;
        }
      }
    }
  }

  if constexpr (LN) {
    __syncthreads();
    if (t < 32) {
      float s = 0.f, ss = 0.f;
      for (int c = 0; c < D; ++c) {
        const float v = Hs[t * 257 + c];
        s += v; ss = fmaf(v, v, ss);
      }
      const float mu = s * (1.0f / D);
      const float var = ss * (1.0f / D) - mu * mu;
      mu_s[t] = mu;
      rs_s[t] = rsqrtf(var + 1e-5f);
    }
    __syncthreads();
    const float g = gamma[t], b = beta[t];
    for (int row = 0; row < 32; ++row) {
      const float val = (Hs[row * 257 + t] - mu_s[row]) * rs_s[row] * g + b;
      Out[(size_t)(row0 + row) * D + t] = (TO)val;
    }
  }
}

// ---------------------------------------------------------------------------
// Attention kernel: one wave per encoder row. Gather 16 neighbor rows into
// VGPRs (lane owns 8 contiguous floats), dot with QT row, softmax over 16,
// weighted neighbor sum -> S (bf16). sb = enc.vvec + c folded into scores.
// ---------------------------------------------------------------------------
__global__ __launch_bounds__(256) void attn_kernel(
    const float* __restrict__ dec, const float* __restrict__ enc,
    const int* __restrict__ idx, const bf16_t* __restrict__ qt,
    const float* __restrict__ vvec, const float* __restrict__ cbuf,
    bf16_t* __restrict__ Sout) {
  const int lane = threadIdx.x & 31;
  const int wv = threadIdx.x >> 5;
  const int n = blockIdx.x * 8 + wv;
  const int col = lane * 8;

  const v8bf qv = *(const v8bf*)(qt + (size_t)n * D + col);
  float qf[8];
#pragma unroll
  for (int i = 0; i < 8; ++i) qf[i] = (float)qv[i];

  // sb = enc_n . vvec  (wave reduce) + c
  const float4* ep = (const float4*)(enc + (size_t)n * D + col);
  const float4 e0 = ep[0], e1 = ep[1];
  const float4* vp = (const float4*)(vvec + col);
  const float4 v0 = vp[0], v1 = vp[1];
  float sb = e0.x * v0.x + e0.y * v0.y + e0.z * v0.z + e0.w * v0.w +
             e1.x * v1.x + e1.y * v1.y + e1.z * v1.z + e1.w * v1.w;
#pragma unroll
  for (int off = 16; off >= 1; off >>= 1) sb += __shfl_xor(sb, off, 32);
  sb += cbuf[0];

  const int* ip = idx + (size_t)n * KNBR;
  float4 nA[KNBR], nB[KNBR];
  float sc[KNBR];
#pragma unroll
  for (int k = 0; k < KNBR; ++k) {
    const int id = ip[k];
    const float4* p = (const float4*)(dec + (size_t)id * D + col);
    nA[k] = p[0]; nB[k] = p[1];
    sc[k] = qf[0] * nA[k].x + qf[1] * nA[k].y + qf[2] * nA[k].z + qf[3] * nA[k].w +
            qf[4] * nB[k].x + qf[5] * nB[k].y + qf[6] * nB[k].z + qf[7] * nB[k].w;
  }
#pragma unroll
  for (int k = 0; k < KNBR; ++k) {
#pragma unroll
    for (int off = 16; off >= 1; off >>= 1) sc[k] += __shfl_xor(sc[k], off, 32);
    sc[k] = (sc[k] + sb) * 0.0625f;  // 1/sqrt(256)
  }

  float m = sc[0];
#pragma unroll
  for (int k = 1; k < KNBR; ++k) m = fmaxf(m, sc[k]);
  float se = 0.f, wgt[KNBR];
#pragma unroll
  for (int k = 0; k < KNBR; ++k) { wgt[k] = __expf(sc[k] - m); se += wgt[k]; }
  const float inv = 1.0f / se;

  float o[8] = {0.f, 0.f, 0.f, 0.f, 0.f, 0.f, 0.f, 0.f};
#pragma unroll
  for (int k = 0; k < KNBR; ++k) {
    const float wk = wgt[k] * inv;
    o[0] = fmaf(wk, nA[k].x, o[0]); o[1] = fmaf(wk, nA[k].y, o[1]);
    o[2] = fmaf(wk, nA[k].z, o[2]); o[3] = fmaf(wk, nA[k].w, o[3]);
    o[4] = fmaf(wk, nB[k].x, o[4]); o[5] = fmaf(wk, nB[k].y, o[5]);
    o[6] = fmaf(wk, nB[k].z, o[6]); o[7] = fmaf(wk, nB[k].w, o[7]);
  }
  v8bf ov;
#pragma unroll
  for (int i = 0; i < 8; ++i) ov[i] = (bf16_t)o[i];
  *(v8bf*)(Sout + (size_t)n * D + col) = ov;
}

// ---------------------------------------------------------------------------
// Host launch
// ---------------------------------------------------------------------------
extern "C" void kernel_launch(void* const* d_in, const int* in_sizes, int n_in,
                              void* d_out, int out_size, void* d_ws, size_t ws_size,
                              hipStream_t stream) {
  (void)in_sizes; (void)n_in; (void)out_size; (void)ws_size;
  const float* dec    = (const float*)d_in[0];
  const float* enc    = (const float*)d_in[1];
  const int*   nidx   = (const int*)d_in[2];
  const float* pos    = (const float*)d_in[3];
  const int*   labels = (const int*)d_in[4];
  const float* Wq = (const float*)d_in[5];
  const float* bq = (const float*)d_in[6];
  const float* Wk = (const float*)d_in[7];
  const float* bk = (const float*)d_in[8];
  const float* Wv = (const float*)d_in[9];
  const float* bv = (const float*)d_in[10];
  const float* W1 = (const float*)d_in[11];
  const float* b1 = (const float*)d_in[12];
  const float* W2 = (const float*)d_in[13];
  const float* b2 = (const float*)d_in[14];
  const float* gamma = (const float*)d_in[15];
  const float* beta  = (const float*)d_in[16];

  char* ws = (char*)d_ws;
  size_t off = 0;
  auto carve = [&](size_t bytes) {
    void* p = ws + off;
    off += (bytes + 255) & ~size_t(255);
    return p;
  };
  bf16_t* enc_bf = (bf16_t*)carve((size_t)N_ENC * D * 2);
  bf16_t* QT_bf  = (bf16_t*)carve((size_t)N_ENC * D * 2);
  bf16_t* S_bf   = (bf16_t*)carve((size_t)N_ENC * D * 2);
  bf16_t* H1_bf  = (bf16_t*)carve((size_t)N_ENC * D * 2);
  bf16_t* Wqk_t  = (bf16_t*)carve((size_t)D * D * 2);
  bf16_t* Wv1_t  = (bf16_t*)carve((size_t)D * D * 2);
  bf16_t* W1b_t  = (bf16_t*)carve((size_t)D * D * 2);
  bf16_t* W2_t   = (bf16_t*)carve((size_t)D * D * 2);
  float*  qb     = (float*)carve(D * 4);
  float*  vvec   = (float*)carve(D * 4);
  float*  b1p    = (float*)carve(D * 4);
  float*  cbuf   = (float*)carve(256);

  float* out = (float*)d_out;

  // --- weight folding / conversions ---
  prep_wqk<<<D, D, 0, stream>>>(Wq, Wk, Wqk_t);
  prep_wv1<<<D, D, 0, stream>>>(Wv, W1, Wv1_t);
  prep_vec<<<4, D, 0, stream>>>(bq, Wk, Wq, bk, bv, W1, b1, qb, vvec, b1p, cbuf);
  prep_cvt<<<256, 256, 0, stream>>>(W1, W2, W1b_t, W2_t);
  prep_enc<<<(N_ENC * D / 4) / 256, 256, 0, stream>>>(enc, enc_bf);

  // K1: QT = enc @ Wqk + qb   -> bf16   (64-row blocks)
  gemm256_kernel<2, false, false, false, bf16_t>
      <<<N_ENC / 64, 256, 0, stream>>>(enc_bf, nullptr, Wqk_t, nullptr, qb,
                                       nullptr, nullptr, QT_bf);

  // K2: attention gather / softmax / weighted neighbor sum -> S (bf16)
  attn_kernel<<<N_ENC / 8, 256, 0, stream>>>(dec, enc, nidx, QT_bf, vvec, cbuf, S_bf);

  // K3: H1 = relu(S @ Wv1 + enc @ W1b + b1p) -> bf16
  gemm256_kernel<2, true, true, false, bf16_t>
      <<<N_ENC / 64, 256, 0, stream>>>(S_bf, enc_bf, Wv1_t, W1b_t, b1p,
                                       nullptr, nullptr, H1_bf);

  // K4: out = LayerNorm(H1 @ W2 + b2) -> f32   (32-row blocks for LDS LN tile)
  gemm256_kernel<1, false, false, true, float>
      <<<N_ENC / 32, 256, 0, stream>>>(H1_bf, nullptr, W2_t, nullptr, b2,
                                       gamma, beta, out);

  // passthrough outputs: encoder_pos then encoder_labels
  unsigned* tail = (unsigned*)(out + (size_t)N_ENC * D);
  copy_u32<<<(N_ENC * 3 + 255) / 256, 256, 0, stream>>>((const unsigned*)pos, tail, N_ENC * 3);
  copy_u32<<<(N_ENC + 255) / 256, 256, 0, stream>>>((const unsigned*)labels, tail + N_ENC * 3, N_ENC);
}
